// Encoder_7662221656403
// MI455X (gfx1250) — matrix operand order repaired and checked by reading, not compiled, not run on hardware
//
#include <hip/hip_runtime.h>
#include <hip/hip_bf16.h>
#include <math.h>

#define B_      2000
#define NLAB    10
#define EMB     500
#define TEMB    50
#define N_AID   200000
#define PAD_AID (N_AID - 1)
#define HCLIP   845
#define INP     5530
#define HID     2765
#define NNEG    60000
#define TOPK    1000
#define INVT    20.0f

// K-dim pads (multiples of 32 elements -> pairs multiple of 16)
#define INP_K2   2768     // 5536/2
#define HIDK_K2  1392     // 2784/2
#define EMB_K2   256      // 512/2
// N-dim pads (multiples of 64)
#define N1_PAD   5568
#define N2_PAD   2816
#define N3_PAD   512
#define NNEG_PAD 60032
#define XLD      5568     // activation buffer leading dim

typedef __attribute__((ext_vector_type(16))) __bf16 v16bf;
typedef __attribute__((ext_vector_type(8)))  float  v8f;

union Frag { v16bf v; unsigned int u[8]; uint4 q[2]; };

static __device__ __forceinline__ unsigned int f2bf(float f) {
    unsigned int u = __float_as_uint(f);
    u = (u + 0x7FFFu + ((u >> 16) & 1u)) >> 16;
    return u & 0xFFFFu;
}

static __device__ __forceinline__ float gelu_exact(float x) {
    return 0.5f * x * (1.0f + erff(x * 0.70710678118654752f));
}

// ---------------------------------------------------------------------------
// 1) Feature build: X (B x XLD f32)
// ---------------------------------------------------------------------------
__global__ void feats_kernel(const int* __restrict__ xy,
                             const float* __restrict__ emb_w,
                             const float* __restrict__ ehw,
                             float* __restrict__ X)
{
    int b = blockIdx.x;
    const int* row = xy + b * 50;
    for (int idx = threadIdx.x; idx < INP; idx += blockDim.x) {
        int j = idx / 553;
        int o = idx - j * 553;
        int aid = row[j];
        int h   = row[10 + j];
        float mask = (aid != PAD_AID) ? 1.0f : 0.0f;
        float v;
        if (o < EMB) {
            v = emb_w[(size_t)aid * EMB + o];
        } else if (o == EMB || o == EMB + 1) {
            float ang = (float)(h % 24) * (6.28318530717958647f / 24.0f);
            v = (o == EMB) ? sinf(ang) : cosf(ang);
        } else if (o == EMB + 2) {
            v = (float)row[20 + j];
        } else {
            int t  = o - (EMB + 3);
            int h1 = min(h + 1, HCLIP);
            int h2 = min(h + 2, HCLIP);
            v = ehw[h * TEMB + t] + ehw[h1 * TEMB + t] + ehw[h2 * TEMB + t];
        }
        X[(size_t)b * XLD + idx] = v * mask;
    }
}

// ---------------------------------------------------------------------------
// 2) BatchNorm + pack A-side to bf16 pair rows: XNp (B x K2pad u32)
// ---------------------------------------------------------------------------
__global__ void bn_pack_kernel(const float* __restrict__ X, int ldx, int Nreal,
                               const float* __restrict__ g,
                               const float* __restrict__ bt,
                               unsigned int* __restrict__ XNp, int K2pad)
{
    int k2 = blockIdx.x * blockDim.x + threadIdx.x;
    if (k2 >= K2pad) return;
    int c0 = 2 * k2, c1 = c0 + 1;
    if (c0 >= Nreal) {
        for (int r = 0; r < B_; r++) XNp[(size_t)r * K2pad + k2] = 0u;
        return;
    }
    bool has1 = (c1 < Nreal);
    float s0 = 0.f, q0 = 0.f, s1 = 0.f, q1 = 0.f;
    for (int r = 0; r < B_; r++) {
        float x0 = X[(size_t)r * ldx + c0]; s0 += x0; q0 += x0 * x0;
        if (has1) { float x1 = X[(size_t)r * ldx + c1]; s1 += x1; q1 += x1 * x1; }
    }
    float m0 = s0 / B_, r0 = rsqrtf(q0 / B_ - m0 * m0 + 1e-5f);
    float g0 = g[c0], b0 = bt[c0];
    float m1 = 0.f, r1 = 0.f, g1 = 0.f, b1v = 0.f;
    if (has1) {
        m1 = s1 / B_; r1 = rsqrtf(q1 / B_ - m1 * m1 + 1e-5f);
        g1 = g[c1]; b1v = bt[c1];
    }
    for (int r = 0; r < B_; r++) {
        float x0 = (X[(size_t)r * ldx + c0] - m0) * r0 * g0 + b0;
        unsigned int lo = f2bf(x0), hi = 0u;
        if (has1) {
            float x1 = (X[(size_t)r * ldx + c1] - m1) * r1 * g1 + b1v;
            hi = f2bf(x1);
        }
        XNp[(size_t)r * K2pad + k2] = lo | (hi << 16);
    }
}

// ---------------------------------------------------------------------------
// 3) Pack fp32 weight (Kreal x Nreal) into B-fragment-order swizzled tiles:
//    out[(((kb*NT) + nt)*32 + lane)*8 + v] holds bf16 pair for
//    k2 = kb*16 + v + (lane>=16 ? 8 : 0), n = nt*16 + (lane&15)
// ---------------------------------------------------------------------------
__global__ void wpack_sw_kernel(const float* __restrict__ W, int Kreal, int Nreal,
                                unsigned int* __restrict__ Wp, int KB, int NT)
{
    size_t idx = (size_t)blockIdx.x * blockDim.x + threadIdx.x;
    size_t tot = (size_t)KB * NT * 256;
    if (idx >= tot) return;
    int v = (int)(idx & 7);
    int l = (int)((idx >> 3) & 31);
    size_t t = idx >> 8;
    int nt = (int)(t % NT);
    int kb = (int)(t / NT);
    int k2 = kb * 16 + v + ((l >> 4) << 3);
    int n  = nt * 16 + (l & 15);
    int k0 = 2 * k2, k1 = k0 + 1;
    unsigned int lo = 0u, hi = 0u;
    if (n < Nreal) {
        if (k0 < Kreal) lo = f2bf(W[(size_t)k0 * Nreal + n]);
        if (k1 < Kreal) hi = f2bf(W[(size_t)k1 * Nreal + n]);
    }
    Wp[idx] = lo | (hi << 16);
}

// ---------------------------------------------------------------------------
// 4) bf16 WMMA GEMM. A: pair-packed rows [m][k2] (lda2). B: swizzled tiles.
//    Each wave: 16 rows x 64 cols (4 WMMA accumulators). Branch-free K-loop,
//    all fragment loads are b128.
// ---------------------------------------------------------------------------
__global__ void gemm_bf16_kernel(const unsigned int* __restrict__ Ap, int lda2,
                                 const unsigned int* __restrict__ Bp, int NT,
                                 float* __restrict__ C, int ldc,
                                 int Mtiles, int Nreal, int K2,
                                 const float* __restrict__ bias, int act)
{
    int lane = threadIdx.x & 31;
    int wave = threadIdx.x >> 5;
    int tm = blockIdx.y * 8 + wave;
    if (tm >= Mtiles) return;                 // wave-uniform
    int hi4 = (lane >> 4) << 2;               // 0 or 4
    int lm  = lane & 15;
    int nt0 = blockIdx.x * 4;

    const unsigned int* arow = Ap + (size_t)(tm * 16 + lm) * lda2;
    // lane's B-fragment base within a tile (32 bytes each)
    const unsigned int* b0 = Bp + ((size_t)nt0 + 0) * 256 + lane * 8;
    const unsigned int* b1 = Bp + ((size_t)nt0 + 1) * 256 + lane * 8;
    const unsigned int* b2 = Bp + ((size_t)nt0 + 2) * 256 + lane * 8;
    const unsigned int* b3 = Bp + ((size_t)nt0 + 3) * 256 + lane * 8;
    size_t kstride = (size_t)NT * 256;        // advance one k-block (16 pairs)

    v8f acc0, acc1, acc2, acc3;
#pragma unroll
    for (int e = 0; e < 8; e++) { acc0[e] = 0.f; acc1[e] = 0.f; acc2[e] = 0.f; acc3[e] = 0.f; }

    for (int k2 = 0; k2 < K2; k2 += 16) {
        Frag a, f0, f1, f2, f3;
        a.q[0] = *(const uint4*)(arow + k2 + 2 * hi4);
        a.q[1] = *(const uint4*)(arow + k2 + 2 * hi4 + 4);
        size_t ko = (size_t)(k2 >> 4) * kstride;
        f0.q[0] = *(const uint4*)(b0 + ko);  f0.q[1] = *(const uint4*)(b0 + ko + 4);
        f1.q[0] = *(const uint4*)(b1 + ko);  f1.q[1] = *(const uint4*)(b1 + ko + 4);
        f2.q[0] = *(const uint4*)(b2 + ko);  f2.q[1] = *(const uint4*)(b2 + ko + 4);
        f3.q[0] = *(const uint4*)(b3 + ko);  f3.q[1] = *(const uint4*)(b3 + ko + 4);
        acc0 = __builtin_amdgcn_wmma_f32_16x16x32_bf16(false, a.v, false, f0.v, (short)0, acc0, false, false);
        acc1 = __builtin_amdgcn_wmma_f32_16x16x32_bf16(false, a.v, false, f1.v, (short)0, acc1, false, false);
        acc2 = __builtin_amdgcn_wmma_f32_16x16x32_bf16(false, a.v, false, f2.v, (short)0, acc2, false, false);
        acc3 = __builtin_amdgcn_wmma_f32_16x16x32_bf16(false, a.v, false, f3.v, (short)0, acc3, false, false);
    }

    int mrow = tm * 16 + ((lane >> 4) << 3);  // + v below
#pragma unroll
    for (int s = 0; s < 4; s++) {
        v8f* ap = (s == 0) ? &acc0 : (s == 1) ? &acc1 : (s == 2) ? &acc2 : &acc3;
        int n = (nt0 + s) * 16 + lm;
        if (n >= Nreal) continue;             // epilogue only
        float bv = bias ? bias[n] : 0.0f;
#pragma unroll
        for (int v = 0; v < 8; v++) {
            float val = (*ap)[v] + bv;
            if (act) val = gelu_exact(val);
            C[(size_t)(mrow + v) * ldc + n] = val;
        }
    }
}

// ---------------------------------------------------------------------------
// 5) Negative row inverse norms
// ---------------------------------------------------------------------------
__global__ void negnorm_kernel(const int* __restrict__ neg_aids,
                               const float* __restrict__ emb_w,
                               float* __restrict__ inv)
{
    int lane = threadIdx.x & 31;
    int r = blockIdx.x * 8 + (threadIdx.x >> 5);
    if (r >= NNEG) return;
    const float* e = emb_w + (size_t)neg_aids[r] * EMB;
    float s = 0.f;
    for (int k = lane; k < EMB; k += 32) { float x = e[k]; s += x * x; }
    for (int off = 16; off >= 1; off >>= 1) s += __shfl_xor(s, off);
    if (lane == 0) inv[r] = rsqrtf(s);
}

// 6) Pack normalized negatives into swizzled B tiles (KB=16, NT=NNEG_PAD/16)
__global__ void negpack_sw_kernel(const int* __restrict__ neg_aids,
                                  const float* __restrict__ emb_w,
                                  const float* __restrict__ inv,
                                  unsigned int* __restrict__ Np, int KB, int NT)
{
    size_t idx = (size_t)blockIdx.x * blockDim.x + threadIdx.x;
    size_t tot = (size_t)KB * NT * 256;
    if (idx >= tot) return;
    int v = (int)(idx & 7);
    int l = (int)((idx >> 3) & 31);
    size_t t = idx >> 8;
    int nt = (int)(t % NT);
    int kb = (int)(t / NT);
    int k2 = kb * 16 + v + ((l >> 4) << 3);
    int n  = nt * 16 + (l & 15);
    int c0 = 2 * k2, c1 = c0 + 1;
    unsigned int lo = 0u, hi = 0u;
    if (n < NNEG) {
        const float* e = emb_w + (size_t)neg_aids[n] * EMB;
        float iv = inv[n];
        if (c0 < EMB) lo = f2bf(e[c0] * iv);
        if (c1 < EMB) hi = f2bf(e[c1] * iv);
    }
    Np[idx] = lo | (hi << 16);
}

// ---------------------------------------------------------------------------
// 7) c_pos per row + packed normalized x_em[:,0]
// ---------------------------------------------------------------------------
__global__ void cpos_kernel(const int* __restrict__ xy,
                            const float* __restrict__ q,       // B x N3_PAD
                            const float* __restrict__ emb_w,
                            const float* __restrict__ lt_w,
                            float* __restrict__ cpos,
                            unsigned int* __restrict__ xem0p)  // B x EMB_K2
{
    int b = blockIdx.x;
    int lane = threadIdx.x;
    const int* row = xy + b * 50;
    const float* qb = q + (size_t)b * N3_PAD;
    float cmin = 1e30f, wsum = 0.f, wcp = 0.f;
    float inv0 = 0.f; int lt0 = 0;
    for (int j = 0; j < NLAB; j++) {
        int yj  = row[30 + j];
        int ltj = row[40 + j];
        const float* lw = lt_w + (size_t)ltj * EMB;
        const float* ye = emb_w + (size_t)yj * EMB;
        float xs = 0.f, ys = 0.f, dt = 0.f;
        for (int k = lane; k < EMB; k += 32) {
            float x = qb[k] + lw[k];
            float y = ye[k];
            xs += x * x; ys += y * y; dt += x * y;
        }
        for (int off = 16; off >= 1; off >>= 1) {
            xs += __shfl_xor(xs, off);
            ys += __shfl_xor(ys, off);
            dt += __shfl_xor(dt, off);
        }
        float invx = rsqrtf(xs);
        float cp = dt * invx * rsqrtf(ys);
        if (j == 0) { inv0 = invx; lt0 = ltj; }
        cmin = fminf(cmin, cp + ((yj == PAD_AID) ? 1.0f : 0.0f));
        wsum += (float)ltj;
        wcp  += cp * (float)ltj;
    }
    if (lane == 0) cpos[b] = 0.5f * (wcp / wsum + cmin);
    const float* lw0 = lt_w + (size_t)lt0 * EMB;
    for (int k2 = lane; k2 < EMB_K2; k2 += 32) {
        int c0 = 2 * k2, c1 = c0 + 1;
        unsigned int lo = 0u, hi = 0u;
        if (c0 < EMB) lo = f2bf((qb[c0] + lw0[c0]) * inv0);
        if (c1 < EMB) hi = f2bf((qb[c1] + lw0[c1]) * inv0);
        xem0p[(size_t)b * EMB_K2 + k2] = lo | (hi << 16);
    }
}

// ---------------------------------------------------------------------------
// 8) Per-row top-1000 (threshold binary search) + log-softmax loss/acc
// ---------------------------------------------------------------------------
__global__ void topkloss_kernel(const float* __restrict__ cneg, int ldn,
                                const float* __restrict__ cpos,
                                float* __restrict__ rloss,
                                float* __restrict__ racc)
{
    __shared__ float sred[256];
    __shared__ int   sredi[256];
    __shared__ float sM, sLo, sHi;
    int tid = threadIdx.x;
    int b = blockIdx.x;
    const float* rowp = cneg + (size_t)b * ldn;

    float mx = -1e30f, mn = 1e30f;
    for (int i = tid; i < NNEG; i += 256) {
        float v = rowp[i]; mx = fmaxf(mx, v); mn = fminf(mn, v);
    }
    sred[tid] = mx; __syncthreads();
    for (int s = 128; s > 0; s >>= 1) {
        if (tid < s) sred[tid] = fmaxf(sred[tid], sred[tid + s]);
        __syncthreads();
    }
    if (tid == 0) sM = sred[0];
    __syncthreads();
    sred[tid] = mn; __syncthreads();
    for (int s = 128; s > 0; s >>= 1) {
        if (tid < s) sred[tid] = fminf(sred[tid], sred[tid + s]);
        __syncthreads();
    }
    if (tid == 0) { sLo = sred[0] - 1.0f; sHi = sM; }
    __syncthreads();
    float M = sM;

    for (int it = 0; it < 48; it++) {
        float mid = 0.5f * (sLo + sHi);
        int c = 0;
        for (int i = tid; i < NNEG; i += 256) c += (rowp[i] > mid) ? 1 : 0;
        sredi[tid] = c; __syncthreads();
        for (int s = 128; s > 0; s >>= 1) {
            if (tid < s) sredi[tid] += sredi[tid + s];
            __syncthreads();
        }
        if (tid == 0) { if (sredi[0] >= TOPK) sLo = mid; else sHi = mid; }
        __syncthreads();
    }
    float thr = sHi;
    float cp = cpos[b];
    float Mall = fmaxf(M, cp);
    int c = 0; float se = 0.f;
    for (int i = tid; i < NNEG; i += 256) {
        float v = rowp[i];
        if (v > thr) { c++; se += expf(INVT * (v - Mall)); }
    }
    sredi[tid] = c; sred[tid] = se; __syncthreads();
    for (int s = 128; s > 0; s >>= 1) {
        if (tid < s) { sredi[tid] += sredi[tid + s]; sred[tid] += sred[tid + s]; }
        __syncthreads();
    }
    if (tid == 0) {
        int k1 = sredi[0];
        float S = sred[0] + (float)(TOPK - k1) * expf(INVT * (thr - Mall))
                + expf(INVT * (cp - Mall));
        rloss[b] = INVT * (Mall - cp) + logf(S);
        racc[b]  = (cp >= M) ? 1.0f : 0.0f;
    }
}

// 9) Fixed-order final reduction -> [loss, acc]
__global__ void final_reduce_kernel(const float* __restrict__ rloss,
                                    const float* __restrict__ racc,
                                    float* __restrict__ out)
{
    __shared__ float sl[256], sa[256];
    int tid = threadIdx.x;
    float l = 0.f, a = 0.f;
    for (int i = tid; i < B_; i += 256) { l += rloss[i]; a += racc[i]; }
    sl[tid] = l; sa[tid] = a; __syncthreads();
    for (int s = 128; s > 0; s >>= 1) {
        if (tid < s) { sl[tid] += sl[tid + s]; sa[tid] += sa[tid + s]; }
        __syncthreads();
    }
    if (tid == 0) { out[0] = sl[0] / (float)B_; out[1] = sa[0] / (float)B_; }
}

// ---------------------------------------------------------------------------
extern "C" void kernel_launch(void* const* d_in, const int* in_sizes, int n_in,
                              void* d_out, int out_size, void* d_ws, size_t ws_size,
                              hipStream_t stream)
{
    (void)in_sizes; (void)n_in; (void)out_size; (void)ws_size;
    const int*   xy     = (const int*)d_in[0];
    const int*   negaid = (const int*)d_in[1];
    const float* emb_w  = (const float*)d_in[2];
    const float* ehw    = (const float*)d_in[3];
    const float* lt_w   = (const float*)d_in[4];
    const float* bn1_g  = (const float*)d_in[5];
    const float* bn1_b  = (const float*)d_in[6];
    const float* w1     = (const float*)d_in[7];
    const float* b1     = (const float*)d_in[8];
    const float* bn2_g  = (const float*)d_in[9];
    const float* bn2_b  = (const float*)d_in[10];
    const float* w2     = (const float*)d_in[11];
    const float* b2     = (const float*)d_in[12];
    const float* bn3_g  = (const float*)d_in[13];
    const float* bn3_b  = (const float*)d_in[14];
    const float* w3     = (const float*)d_in[15];
    const float* b3     = (const float*)d_in[16];

    char* ws = (char*)d_ws;
    size_t off = 0;
    auto take = [&](size_t bytes) { size_t o = off; off += (bytes + 255) & ~(size_t)255; return o; };
    float*        X     = (float*)(ws + take((size_t)B_ * XLD * 4));
    unsigned int* XNp   = (unsigned int*)(ws + take((size_t)B_ * INP_K2 * 4));
    unsigned int* Wp    = (unsigned int*)(ws + take((size_t)INP_K2 * N1_PAD * 4));
    float*        q     = (float*)(ws + take((size_t)B_ * N3_PAD * 4));
    float*        ninv  = (float*)(ws + take((size_t)NNEG * 4));
    unsigned int* Np    = (unsigned int*)(ws + take((size_t)EMB_K2 * NNEG_PAD * 4));
    unsigned int* xem0p = (unsigned int*)(ws + take((size_t)B_ * EMB_K2 * 4));
    float*        cneg  = (float*)(ws + take((size_t)B_ * NNEG_PAD * 4));
    float*        cposb = (float*)(ws + take((size_t)B_ * 4));
    float*        rloss = (float*)(ws + take((size_t)B_ * 4));
    float*        racc  = (float*)(ws + take((size_t)B_ * 4));

    feats_kernel<<<B_, 256, 0, stream>>>(xy, emb_w, ehw, X);

    // layer 1
    bn_pack_kernel<<<(INP_K2 + 255) / 256, 256, 0, stream>>>(X, XLD, INP, bn1_g, bn1_b, XNp, INP_K2);
    wpack_sw_kernel<<<(int)(((size_t)(INP_K2 / 16) * (N1_PAD / 16) * 256 + 255) / 256), 256, 0, stream>>>(
        w1, INP, INP, Wp, INP_K2 / 16, N1_PAD / 16);
    gemm_bf16_kernel<<<dim3(N1_PAD / 64, 16), 256, 0, stream>>>(
        XNp, INP_K2, Wp, N1_PAD / 16, X, XLD, B_ / 16, INP, INP_K2, b1, 1);

    // layer 2
    bn_pack_kernel<<<(INP_K2 + 255) / 256, 256, 0, stream>>>(X, XLD, INP, bn2_g, bn2_b, XNp, INP_K2);
    wpack_sw_kernel<<<(int)(((size_t)(INP_K2 / 16) * (N2_PAD / 16) * 256 + 255) / 256), 256, 0, stream>>>(
        w2, INP, HID, Wp, INP_K2 / 16, N2_PAD / 16);
    gemm_bf16_kernel<<<dim3(N2_PAD / 64, 16), 256, 0, stream>>>(
        XNp, INP_K2, Wp, N2_PAD / 16, X, N2_PAD, B_ / 16, HID, INP_K2, b2, 1);

    // layer 3 -> q
    bn_pack_kernel<<<(HIDK_K2 + 255) / 256, 256, 0, stream>>>(X, N2_PAD, HID, bn3_g, bn3_b, XNp, HIDK_K2);
    wpack_sw_kernel<<<(int)(((size_t)(HIDK_K2 / 16) * (N3_PAD / 16) * 256 + 255) / 256), 256, 0, stream>>>(
        w3, HID, EMB, Wp, HIDK_K2 / 16, N3_PAD / 16);
    gemm_bf16_kernel<<<dim3(N3_PAD / 64, 16), 256, 0, stream>>>(
        XNp, HIDK_K2, Wp, N3_PAD / 16, q, N3_PAD, B_ / 16, EMB, HIDK_K2, b3, 0);

    // negatives
    negnorm_kernel<<<(NNEG + 7) / 8, 256, 0, stream>>>(negaid, emb_w, ninv);
    negpack_sw_kernel<<<(int)(((size_t)(EMB_K2 / 16) * (NNEG_PAD / 16) * 256 + 255) / 256), 256, 0, stream>>>(
        negaid, emb_w, ninv, Np, EMB_K2 / 16, NNEG_PAD / 16);

    cpos_kernel<<<B_, 32, 0, stream>>>(xy, q, emb_w, lt_w, cposb, xem0p);

    // similarity GEMM (2000 x 512) x (512 x 60032)
    gemm_bf16_kernel<<<dim3(NNEG_PAD / 64, 16), 256, 0, stream>>>(
        xem0p, EMB_K2, Np, NNEG_PAD / 16, cneg, NNEG_PAD, B_ / 16, NNEG, EMB_K2, nullptr, 0);

    topkloss_kernel<<<B_, 256, 0, stream>>>(cneg, NNEG_PAD, cposb, rloss, racc);
    final_reduce_kernel<<<1, 256, 0, stream>>>(rloss, racc, (float*)d_out);
}
